// TrgAwaDecoder_48533130445096
// MI455X (gfx1250) — matrix operand order repaired and checked by reading, not compiled, not run on hardware
//
#include <hip/hip_runtime.h>

// ---------------------------------------------------------------------------
// TrgAwaDecoder fused kernels for gfx1250 (MI455X, wave32, WMMA)
//   pass 1 : LayerNorm(src)->Kh(f16), LayerNorm(trg)->Qh(f16)
//   pass 1b: LDS-tiled transpose Kh -> VhT ([B][D][S], coalesced both sides)
//   pass 2 : flash attention M=32 (QK^T -> mask -> online softmax -> PV)
//            + residual + final LayerNorm, all WMMA f16->f32
// ---------------------------------------------------------------------------

static constexpr int Bn = 4;
static constexpr int Sn = 2048;
static constexpr int Dn = 1024;
static constexpr float EPSV = 1e-5f;
static constexpr float NEGV = -1e9f;
static constexpr int WAVES = 8;       // 256 threads / 32
static constexpr int DSLICE = 128;    // D columns owned per wave

typedef _Float16 h4  __attribute__((ext_vector_type(4)));
typedef _Float16 h8  __attribute__((ext_vector_type(8)));
typedef _Float16 v16h __attribute__((ext_vector_type(16)));
typedef float    v8f  __attribute__((ext_vector_type(8)));

union V16 { v16h v; h8 h[2]; };

// ---------------------------------------------------------------------------
// Kernel 1: row-wise LayerNorm, cast to f16 (row major only).
//   rows [0, B*S)      : trg -> Qh
//   rows [B*S, 2*B*S)  : src -> Kh
// ---------------------------------------------------------------------------
__global__ __launch_bounds__(256)
void ln_cast_kernel(const float* __restrict__ src, const float* __restrict__ trg,
                    const float* __restrict__ gamma, const float* __restrict__ beta,
                    _Float16* __restrict__ Qh, _Float16* __restrict__ Kh) {
  const int row   = blockIdx.x;               // 0 .. 2*B*S-1
  const bool is_src = row >= Bn * Sn;
  const int r     = is_src ? (row - Bn * Sn) : row;
  const float* x  = (is_src ? src : trg) + (size_t)r * Dn;
  const int tid   = threadIdx.x;              // 256 threads, 4 elements each
  const int d0    = tid * 4;

  float4 v = *(const float4*)(x + d0);
  float s1 = v.x + v.y + v.z + v.w;
  float s2 = v.x * v.x + v.y * v.y + v.z * v.z + v.w * v.w;
#pragma unroll
  for (int off = 16; off > 0; off >>= 1) {
    s1 += __shfl_xor(s1, off, 32);
    s2 += __shfl_xor(s2, off, 32);
  }
  __shared__ float ws1[WAVES], ws2[WAVES];
  __shared__ float mu_s, rs_s;
  const int w = tid >> 5, lane = tid & 31;
  if (lane == 0) { ws1[w] = s1; ws2[w] = s2; }
  __syncthreads();
  if (tid == 0) {
    float a = 0.f, q = 0.f;
#pragma unroll
    for (int i = 0; i < WAVES; ++i) { a += ws1[i]; q += ws2[i]; }
    float mu = a * (1.0f / Dn);
    float var = q * (1.0f / Dn) - mu * mu;
    mu_s = mu;
    rs_s = rsqrtf(var + EPSV);
  }
  __syncthreads();
  const float mu = mu_s, rs = rs_s;

  h4 hv;
#pragma unroll
  for (int j = 0; j < 4; ++j) {
    const int d = d0 + j;
    const float e = (j == 0 ? v.x : j == 1 ? v.y : j == 2 ? v.z : v.w);
    hv[j] = (_Float16)((e - mu) * rs * gamma[d] + beta[d]);
  }
  if (!is_src) *(h4*)(Qh + (size_t)r * Dn + d0) = hv;
  else         *(h4*)(Kh + (size_t)r * Dn + d0) = hv;
}

// ---------------------------------------------------------------------------
// Kernel 1b: 32x32 LDS-tiled transpose  Kh[B][S][D] -> VhT[B][D][S].
// ---------------------------------------------------------------------------
__global__ __launch_bounds__(256)
void transpose_kernel(const _Float16* __restrict__ Kh, _Float16* __restrict__ VhT) {
  constexpr int TD = Dn / 32;   // 32 tiles along D
  constexpr int TS = Sn / 32;   // 64 tiles along S
  int t = blockIdx.x;
  const int b = t / (TD * TS);
  t %= TD * TS;
  const int s0 = (t / TD) * 32;
  const int d0 = (t % TD) * 32;

  __shared__ _Float16 tile[32][33];   // [d_local][s_local], padded
  const int c = threadIdx.x & 31;
  const int r = threadIdx.x >> 5;     // 0..7
#pragma unroll
  for (int rr = r; rr < 32; rr += 8)
    tile[c][rr] = Kh[((size_t)(b * Sn + s0 + rr)) * Dn + d0 + c];   // coalesced read
  __syncthreads();
#pragma unroll
  for (int rr = r; rr < 32; rr += 8)
    VhT[((size_t)(b * Dn + d0 + rr)) * Sn + s0 + c] = tile[rr][c];  // coalesced write
}

// ---------------------------------------------------------------------------
// Kernel 2: fused flash attention (M=32) + residual + final LayerNorm.
// grid = B * (S/32) blocks of 256 threads (8 waves).
// Wave w owns D-slice [w*128, w*128+128) for split-K of QK^T and for PV out.
// Per s-step: A) QK WMMA, store partials; B) all 8 waves reduce+mask one
// slice each; C) waves 0,1 run online softmax per M-subtile; D) PV WMMA.
// ---------------------------------------------------------------------------
__global__ __launch_bounds__(256)
void flash_attn_kernel(const float* __restrict__ src, const float* __restrict__ kpm,
                       const float* __restrict__ mmask,
                       const float* __restrict__ gamma, const float* __restrict__ beta,
                       const _Float16* __restrict__ Qh, const _Float16* __restrict__ Kh,
                       const _Float16* __restrict__ VhT, float* __restrict__ out) {
  constexpr int MT = Sn / 32;            // 64 m-tiles per batch
  const int bid  = blockIdx.x;
  const int b    = bid / MT;
  const int m0   = (bid % MT) * 32;      // 32 trg rows per block
  const int tid  = threadIdx.x;
  const int w    = tid >> 5;
  const int lane = tid & 31;
  const int half = lane >> 4;            // 0: lanes 0-15, 1: lanes 16-31
  const int ln   = lane & 15;
  const int dW   = w * DSLICE;

  __shared__ float    red[WAVES][2][2][8][32];  // [wave][mt][j][vgpr][lane] 32 KB
  __shared__ float    sred[2][2][8][32];        // reduced+masked scores (4 KB)
  __shared__ _Float16 Plds[2][16][32];          // P tiles, row major (2 KB)
  __shared__ float    alphaL[2][16];
  __shared__ float    lsum[2][16];
  __shared__ float    rowstat[WAVES][32][2];
  __shared__ float    musig[32][2];

  // --- Q A-fragments (16x32 f16 layout) for both M-subtiles -----------------
  V16 qa[2][4];
#pragma unroll
  for (int mt = 0; mt < 2; ++mt) {
    const _Float16* qbase = Qh + ((size_t)(b * Sn + m0 + mt * 16 + ln)) * Dn;
#pragma unroll
    for (int kc = 0; kc < 4; ++kc) {
      const int k0 = dW + kc * 32 + half * 16;
      qa[mt][kc].h[0] = *(const h8*)(qbase + k0);
      qa[mt][kc].h[1] = *(const h8*)(qbase + k0 + 8);
    }
  }

  v8f acc[2][8];
#pragma unroll
  for (int mt = 0; mt < 2; ++mt)
#pragma unroll
    for (int c = 0; c < 8; ++c)
      acc[mt][c] = (v8f){0.f, 0.f, 0.f, 0.f, 0.f, 0.f, 0.f, 0.f};

  float mrow[8], lrow[8];                // online-softmax state (waves 0,1 only)
#pragma unroll
  for (int i = 0; i < 8; ++i) { mrow[i] = -3.0e38f; lrow[i] = 0.f; }

  const _Float16* kbase = Kh + ((size_t)b * Sn) * Dn;
  const _Float16* vbase = VhT + ((size_t)b * Dn) * Sn;

  // reduction slice owned by this wave in phase B
  const int rmt = w & 1, rj = (w >> 1) & 1, rib = (w >> 2) * 4;

  for (int s0 = 0; s0 < Sn; s0 += 32) {
    // ==== phase A: QK^T partials over this wave's 128-wide K slice ==========
    // hoist all 8 K B-fragments so the loads form one deep clause
    V16 bk[2][4];
#pragma unroll
    for (int j = 0; j < 2; ++j) {
      const _Float16* krow = kbase + (size_t)(s0 + j * 16 + ln) * Dn;
#pragma unroll
      for (int kc = 0; kc < 4; ++kc) {
        const int k0 = dW + kc * 32 + half * 16;   // B 32x16: lane=N, 16 contiguous K
        bk[j][kc].h[0] = *(const h8*)(krow + k0);
        bk[j][kc].h[1] = *(const h8*)(krow + k0 + 8);
      }
    }
    v8f cs[2][2];
#pragma unroll
    for (int mt = 0; mt < 2; ++mt)
#pragma unroll
      for (int j = 0; j < 2; ++j)
        cs[mt][j] = (v8f){0.f, 0.f, 0.f, 0.f, 0.f, 0.f, 0.f, 0.f};
#pragma unroll
    for (int j = 0; j < 2; ++j)
#pragma unroll
      for (int kc = 0; kc < 4; ++kc)
#pragma unroll
        for (int mt = 0; mt < 2; ++mt)             // K frag shared by both M-subtiles
          cs[mt][j] = __builtin_amdgcn_wmma_f32_16x16x32_f16(
              false, qa[mt][kc].v, false, bk[j][kc].v, (short)0, cs[mt][j], false, false);
#pragma unroll
    for (int mt = 0; mt < 2; ++mt)
#pragma unroll
      for (int j = 0; j < 2; ++j)
#pragma unroll
        for (int i = 0; i < 8; ++i) red[w][mt][j][i][lane] = cs[mt][j][i];

    // ---- V B-fragments (shared by both M-subtiles; independent of P) -------
    V16 bv[8];
#pragma unroll
    for (int c = 0; c < 8; ++c) {
      const _Float16* vrow = vbase + (size_t)(dW + c * 16 + ln) * Sn + s0 + half * 16;
      bv[c].h[0] = *(const h8*)(vrow);
      bv[c].h[1] = *(const h8*)(vrow + 8);
    }
    if (s0 + 32 < Sn) {  // prefetch next K and V tiles (global_prefetch_b8)
      __builtin_prefetch(kbase + (size_t)(s0 + 32 + ln) * Dn + dW, 0, 1);
      __builtin_prefetch(vbase + (size_t)(dW + ln) * Sn + s0 + 32, 0, 1);
    }

    __syncthreads();   // red[] visible

    // ==== phase B: all 8 waves reduce + scale + mask one slice each =========
    {
#pragma unroll
      for (int ii = 0; ii < 4; ++ii) {
        const int i = rib + ii;
        float v = 0.f;
#pragma unroll
        for (int ww = 0; ww < WAVES; ++ww) v += red[ww][rmt][rj][i][lane];
        v *= (1.0f / (float)Dn);
        const int t  = m0 + rmt * 16 + i + half * 8;
        const int sg = s0 + rj * 16 + ln;
        if (kpm[b * Sn + sg] == 0.0f) v = NEGV;
        if (mmask[(size_t)t * Sn + sg] == 0.0f) v = NEGV;
        sred[rmt][rj][i][lane] = v;
      }
    }
    __syncthreads();   // sred[] visible

    // ==== phase C: waves 0,1 run the online softmax for one M-subtile =======
    if (w < 2) {
      const int mt = w;
#pragma unroll
      for (int i = 0; i < 8; ++i) {
        const float pv0 = sred[mt][0][i][lane];
        const float pv1 = sred[mt][1][i][lane];
        float tmax = fmaxf(pv0, pv1);
#pragma unroll
        for (int off = 1; off < 16; off <<= 1)
          tmax = fmaxf(tmax, __shfl_xor(tmax, off, 32));
        const float mnew  = fmaxf(mrow[i], tmax);
        const float alpha = __expf(mrow[i] - mnew);
        const float p0 = __expf(pv0 - mnew);
        const float p1 = __expf(pv1 - mnew);
        Plds[mt][i + half * 8][ln]      = (_Float16)p0;
        Plds[mt][i + half * 8][16 + ln] = (_Float16)p1;
        float ssum = p0 + p1;
#pragma unroll
        for (int off = 1; off < 16; off <<= 1)
          ssum += __shfl_xor(ssum, off, 32);
        lrow[i] = lrow[i] * alpha + ssum;
        mrow[i] = mnew;
        if (ln == 0) alphaL[mt][i + half * 8] = alpha;
      }
    }
    __syncthreads();   // Plds/alphaL visible
    // (no trailing barrier needed: next iteration's writers of red/sred/Plds
    //  are separated from this iteration's readers by the barriers above)

    // ==== phase D: all waves rescale O, load P A-fragments, PV WMMA =========
#pragma unroll
    for (int mt = 0; mt < 2; ++mt) {
      float a8[8];
#pragma unroll
      for (int i = 0; i < 8; ++i) a8[i] = alphaL[mt][i + half * 8];
#pragma unroll
      for (int c = 0; c < 8; ++c)
#pragma unroll
        for (int i = 0; i < 8; ++i) acc[mt][c][i] *= a8[i];

      V16 pf;
      {
        const _Float16* prow = &Plds[mt][ln][0];
        const int k0 = half * 16;
        pf.h[0] = *(const h8*)(prow + k0);
        pf.h[1] = *(const h8*)(prow + k0 + 8);
      }
#pragma unroll
      for (int c = 0; c < 8; ++c)
        acc[mt][c] = __builtin_amdgcn_wmma_f32_16x16x32_f16(
            false, pf.v, false, bv[c].v, (short)0, acc[mt][c], false, false);
    }
  }

  // ---- epilogue: 1/l normalize, + residual, fused final LayerNorm ----------
  __syncthreads();
  if (w < 2 && ln == 0) {
#pragma unroll
    for (int i = 0; i < 8; ++i) lsum[w][i + half * 8] = lrow[i];
  }
  __syncthreads();

  float linv[2][8];
#pragma unroll
  for (int mt = 0; mt < 2; ++mt)
#pragma unroll
    for (int i = 0; i < 8; ++i)
      linv[mt][i] = 1.0f / fmaxf(lsum[mt][i + half * 8], 1e-20f);

#pragma unroll
  for (int mt = 0; mt < 2; ++mt)
#pragma unroll
    for (int i = 0; i < 8; ++i) {
      const int t = m0 + mt * 16 + i + half * 8;
      const float* srow = src + ((size_t)(b * Sn + t)) * Dn;
      float s1 = 0.f, s2 = 0.f;
#pragma unroll
      for (int c = 0; c < 8; ++c) {
        const int d = dW + c * 16 + ln;
        const float x = acc[mt][c][i] * linv[mt][i] + srow[d];
        acc[mt][c][i] = x;
        s1 += x;
        s2 += x * x;
      }
#pragma unroll
      for (int off = 1; off < 16; off <<= 1) {
        s1 += __shfl_xor(s1, off, 32);
        s2 += __shfl_xor(s2, off, 32);
      }
      if (ln == 0) {
        rowstat[w][mt * 16 + i + half * 8][0] = s1;
        rowstat[w][mt * 16 + i + half * 8][1] = s2;
      }
    }
  __syncthreads();
  if (tid < 32) {
    float a = 0.f, q = 0.f;
#pragma unroll
    for (int ww = 0; ww < WAVES; ++ww) { a += rowstat[ww][tid][0]; q += rowstat[ww][tid][1]; }
    const float mu = a * (1.0f / Dn);
    const float var = q * (1.0f / Dn) - mu * mu;
    musig[tid][0] = mu;
    musig[tid][1] = rsqrtf(var + EPSV);
  }
  __syncthreads();

#pragma unroll
  for (int mt = 0; mt < 2; ++mt)
#pragma unroll
    for (int i = 0; i < 8; ++i) {
      const int t  = m0 + mt * 16 + i + half * 8;
      const float mu = musig[mt * 16 + i + half * 8][0];
      const float rs = musig[mt * 16 + i + half * 8][1];
      float* orow = out + ((size_t)(b * Sn + t)) * Dn;
#pragma unroll
      for (int c = 0; c < 8; ++c) {
        const int d = dW + c * 16 + ln;
        orow[d] = (acc[mt][c][i] - mu) * rs * gamma[d] + beta[d];
      }
    }
}

// ---------------------------------------------------------------------------
extern "C" void kernel_launch(void* const* d_in, const int* in_sizes, int n_in,
                              void* d_out, int out_size, void* d_ws, size_t ws_size,
                              hipStream_t stream) {
  const float* src   = (const float*)d_in[0];
  const float* trg   = (const float*)d_in[1];
  const float* kpm   = (const float*)d_in[2];   // (B,1,S)
  const float* mmask = (const float*)d_in[3];   // (S,S)
  const float* gamma = (const float*)d_in[4];
  const float* beta  = (const float*)d_in[5];
  float* out = (float*)d_out;

  // workspace: Qh (16 MB) | Kh (16 MB) | VhT (16 MB)
  const size_t half_bytes = (size_t)Bn * Sn * Dn * sizeof(_Float16);
  char* ws = (char*)d_ws;
  _Float16* Qh  = (_Float16*)(ws);
  _Float16* Kh  = (_Float16*)(ws + half_bytes);
  _Float16* VhT = (_Float16*)(ws + 2 * half_bytes);

  ln_cast_kernel<<<2 * Bn * Sn, 256, 0, stream>>>(src, trg, gamma, beta, Qh, Kh);
  transpose_kernel<<<Bn * (Sn / 32) * (Dn / 32), 256, 0, stream>>>(Kh, VhT);
  flash_attn_kernel<<<Bn * (Sn / 32), 256, 0, stream>>>(src, kpm, mmask, gamma, beta,
                                                        Qh, Kh, VhT, out);
  (void)in_sizes; (void)n_in; (void)out_size; (void)ws_size;
}
